// TopKSAE_25795573579942
// MI455X (gfx1250) — matrix operand order repaired and checked by reading, not compiled, not run on hardware
//
#include <hip/hip_runtime.h>
#include <stdint.h>

#define BATCH   8192
#define DMODEL  768
#define DSAE    16384
#define KTOP    64

#define BM 128
#define BN 128
#define BK 32
#define LDA 40   // padded LDS row (bf16 elems): conflict-free b128 reads
#define LDB 40

typedef __attribute__((ext_vector_type(16))) __bf16 v16bf;
typedef __attribute__((ext_vector_type(8)))  __bf16 v8bf;
typedef __attribute__((ext_vector_type(8)))  float  v8f;

union FragAB { v16bf v; v8bf h[2]; };

// Tensor Data Mover availability (device pass only; host pass sees 0).
#if defined(__has_builtin)
#  if __has_builtin(__builtin_amdgcn_tensor_load_to_lds)
#    define HAVE_TDM 1
#  endif
#endif
#ifndef HAVE_TDM
#  define HAVE_TDM 0
#endif

__device__ __forceinline__ uint16_t cvt_bf16_bits(float f) {
  union { float f; uint32_t u; } a; a.f = f;
  uint32_t r = a.u + 0x7FFFu + ((a.u >> 16) & 1u);   // round-to-nearest-even
  return (uint16_t)(r >> 16);
}
__device__ __forceinline__ __bf16 cvt_bf16(float f) {
  union { uint16_t s; __bf16 h; } b; b.s = cvt_bf16_bits(f);
  return b.h;
}

// gfx1250 async copy: 16 bytes/lane, global -> LDS, tracked by ASYNCcnt.
__device__ __forceinline__ void async_ld16(uint32_t lds_byte_off, const void* gaddr) {
  asm volatile("global_load_async_to_lds_b128 %0, %1, off"
               :: "v"(lds_byte_off), "v"((uint64_t)(uintptr_t)gaddr)
               : "memory");
}
__device__ __forceinline__ uint32_t lds_off(const void* p) {
  // generic LDS pointer: addr[31:0] is the LDS offset (aperture in high bits)
  return (uint32_t)(uintptr_t)p;
}

#if HAVE_TDM
typedef uint32_t v4u  __attribute__((ext_vector_type(4)));
typedef int      v8i_ __attribute__((ext_vector_type(8)));
typedef int      v4i_ __attribute__((ext_vector_type(4)));

// One TDM descriptor: 2-D bf16 tile (BK x BM rows) global->LDS, with HW LDS
// padding of 4 DWORDs every 16 DWORDs (64B row -> 80B LDS row == LDA/LDB=40).
__device__ __forceinline__ void tdm_load_tile(uint32_t ldsaddr, const void* gtile,
                                              uint32_t tensor_rows) {
  const uint64_t ga = (uint64_t)(uintptr_t)gtile;
  v4u g0 = { 1u,                                            // count=1, user mode
             ldsaddr,                                       // lds_addr
             (uint32_t)(ga & 0xFFFFFFFFu),                  // global_addr[31:0]
             (uint32_t)((ga >> 32) & 0x01FFFFFFu) | (2u << 30) }; // addr[56:32]|type=2
  const uint32_t d0 = (1u << 16)    // data_size = 2 bytes
                    | (1u << 20)    // pad_enable
                    | (3u << 22)    // pad_interval: 16 DWORDs (64B)
                    | (3u << 25);   // pad_amount:   4 DWORDs (16B)
  const uint32_t td0 = DMODEL, td1 = tensor_rows;
  v8i_ g1 = { (int)d0,
              (int)((td0 & 0xFFFFu) << 16),                             // abar=0 | td0.lo
              (int)(((td0 >> 16) & 0xFFFFu) | ((td1 & 0xFFFFu) << 16)), // td0.hi | td1.lo
              (int)(((td1 >> 16) & 0xFFFFu) | ((uint32_t)BK << 16)),    // td1.hi | tile0=32
              (int)(uint32_t)BM,                                        // tile1=128 | tile2=0
              (int)(uint32_t)DMODEL,                                    // stride0[31:0]
              0,                                                        // stride0.hi|stride1.lo
              0 };
  v4i_ gz = { 0, 0, 0, 0 };
#if __clang_major__ >= 23
  v8i_ gz8 = { 0, 0, 0, 0, 0, 0, 0, 0 };
  __builtin_amdgcn_tensor_load_to_lds(g0, g1, gz, gz, gz8, 0);
#else
  __builtin_amdgcn_tensor_load_to_lds(g0, g1, gz, gz, 0);
#endif
}
#endif // HAVE_TDM

// ---------------------------------------------------------------------------
// Kernel 0a: x (fp32) -> bf16, same row-major layout.
// ---------------------------------------------------------------------------
__global__ __launch_bounds__(256)
void k_cvt_x(const float* __restrict__ in, uint16_t* __restrict__ out) {
  size_t i = ((size_t)blockIdx.x * 256 + threadIdx.x) * 8;
  float4 f0 = *(const float4*)(in + i);
  float4 f1 = *(const float4*)(in + i + 4);
  union { uint16_t s[8]; uint4 u; } p;
  p.s[0] = cvt_bf16_bits(f0.x); p.s[1] = cvt_bf16_bits(f0.y);
  p.s[2] = cvt_bf16_bits(f0.z); p.s[3] = cvt_bf16_bits(f0.w);
  p.s[4] = cvt_bf16_bits(f1.x); p.s[5] = cvt_bf16_bits(f1.y);
  p.s[6] = cvt_bf16_bits(f1.z); p.s[7] = cvt_bf16_bits(f1.w);
  *(uint4*)(out + i) = p.u;
}

// ---------------------------------------------------------------------------
// Kernel 0b: W_enc [k][n] fp32 -> wt [n][k] bf16 (transposed), 64x64 LDS tiles.
// ---------------------------------------------------------------------------
__global__ __launch_bounds__(256)
void k_cvt_wT(const float* __restrict__ W, uint16_t* __restrict__ wt) {
  __shared__ __align__(16) uint16_t Tl[64 * 72];
  const int t  = threadIdx.x;
  const int n0 = blockIdx.x * 64;
  const int k0 = blockIdx.y * 64;
  {
    const int kr = t >> 4;
    const int nc = (t & 15) << 2;
    #pragma unroll
    for (int it = 0; it < 4; ++it) {
      const int k = kr + it * 16;
      const float4 f = *(const float4*)(W + (size_t)(k0 + k) * DSAE + n0 + nc);
      Tl[(nc + 0) * 72 + k] = cvt_bf16_bits(f.x);
      Tl[(nc + 1) * 72 + k] = cvt_bf16_bits(f.y);
      Tl[(nc + 2) * 72 + k] = cvt_bf16_bits(f.z);
      Tl[(nc + 3) * 72 + k] = cvt_bf16_bits(f.w);
    }
  }
  __syncthreads();
  {
    const int nl = t >> 2;
    const int kc = (t & 3) << 4;
    const uint4 a = *(const uint4*)(&Tl[nl * 72 + kc]);
    const uint4 b = *(const uint4*)(&Tl[nl * 72 + kc + 8]);
    uint16_t* dst = wt + (size_t)(n0 + nl) * DMODEL + k0 + kc;
    *(uint4*)dst       = a;
    *(uint4*)(dst + 8) = b;
  }
}

// ---------------------------------------------------------------------------
// Kernel 1 (fast path): bf16 GEMM, double-buffered TDM (or async-to-LDS)
// staging. Block 128x128, 8 waves (2Mx4N). z <- relu(x@W_enc + b_enc).
// ---------------------------------------------------------------------------
__global__ __launch_bounds__(256)
void sae_encode_wmma_async(const uint16_t* __restrict__ xbu,
                           const uint16_t* __restrict__ wtu,
                           const float* __restrict__ b_enc,
                           float* __restrict__ z) {
  __shared__ __align__(16) __bf16 Abuf[2][BM * LDA];   // [m][k]
  __shared__ __align__(16) __bf16 Bbuf[2][BN * LDB];   // [n][k]
  const __bf16* xb = (const __bf16*)xbu;
  const __bf16* wt = (const __bf16*)wtu;

  const int tid  = threadIdx.x;
  const int lane = tid & 31;
  const int wid  = tid >> 5;
  const int wm   = wid >> 2;
  const int wn   = wid & 3;
  const int half = lane >> 4;
  const int ln   = lane & 15;
  const int m0   = blockIdx.y * BM;
  const int n0   = blockIdx.x * BN;

#if HAVE_TDM
  // 2 TDM descriptor loads per stage, issued by wave 0 only (TENSORcnt).
  auto issue = [&](int buf, int k0) {
    tdm_load_tile(lds_off(&Abuf[buf][0]), xb + (size_t)m0 * DMODEL + k0, BATCH);
    tdm_load_tile(lds_off(&Bbuf[buf][0]), wt + (size_t)n0 * DMODEL + k0, DSAE);
  };
#else
  // 4 async b128 instructions per wave per stage (A:2 + B:2).
  auto issue = [&](int buf, int k0) {
    #pragma unroll
    for (int it = 0; it < 2; ++it) {
      const int c = tid + it * 256;      // 16B chunk id, 0..511
      const int r = c >> 2;              // tile row 0..127
      const int q = (c & 3) << 3;        // element offset 0,8,16,24
      async_ld16(lds_off(&Abuf[buf][r * LDA + q]),
                 xb + (size_t)(m0 + r) * DMODEL + k0 + q);
      async_ld16(lds_off(&Bbuf[buf][r * LDB + q]),
                 wt + (size_t)(n0 + r) * DMODEL + k0 + q);
    }
  };
#endif

  v8f acc[4][2];
  #pragma unroll
  for (int i = 0; i < 4; ++i)
    #pragma unroll
    for (int j = 0; j < 2; ++j)
      #pragma unroll
      for (int e = 0; e < 8; ++e) acc[i][j][e] = 0.0f;

#if HAVE_TDM
  if (wid == 0) issue(0, 0);
#else
  issue(0, 0);
#endif

  const int NSTEP = DMODEL / BK;   // 24
  for (int s = 0; s < NSTEP; ++s) {
    const int buf = s & 1;
#if HAVE_TDM
    if (s + 1 < NSTEP) {
      if (wid == 0) {
        issue(buf ^ 1, (s + 1) * BK);
        __builtin_amdgcn_s_wait_tensorcnt(2);   // current buf landed
      }
    } else {
      if (wid == 0) __builtin_amdgcn_s_wait_tensorcnt(0);
    }
#else
    if (s + 1 < NSTEP) {
      issue(buf ^ 1, (s + 1) * BK);
      asm volatile("s_wait_asynccnt 0x4" ::: "memory");  // current buf landed
    } else {
      asm volatile("s_wait_asynccnt 0x0" ::: "memory");
    }
#endif
    __syncthreads();

    FragAB fa[4], fb[2];
    #pragma unroll
    for (int tm = 0; tm < 4; ++tm) {
      const __bf16* ap = &Abuf[buf][(wm * 64 + tm * 16 + ln) * LDA + (half << 3)];
      fa[tm].h[0] = *(const v8bf*)ap;         // K = 8*half + 0..7
      fa[tm].h[1] = *(const v8bf*)(ap + 16);  // K = 8*half + 16..23
    }
    #pragma unroll
    for (int tn = 0; tn < 2; ++tn) {
      const __bf16* bp = &Bbuf[buf][(wn * 32 + tn * 16 + ln) * LDB + (half << 3)];
      fb[tn].h[0] = *(const v8bf*)bp;         // K = 8*half + 0..7
      fb[tn].h[1] = *(const v8bf*)(bp + 16);  // K = 8*half + 16..23
    }
    #pragma unroll
    for (int tm = 0; tm < 4; ++tm)
      #pragma unroll
      for (int tn = 0; tn < 2; ++tn)
        acc[tm][tn] = __builtin_amdgcn_wmma_f32_16x16x32_bf16(
            false, fa[tm].v, false, fb[tn].v, (short)0, acc[tm][tn], false, false);
    __syncthreads();   // all waves done reading buf before it is overwritten
  }

  #pragma unroll
  for (int tm = 0; tm < 4; ++tm) {
    #pragma unroll
    for (int tn = 0; tn < 2; ++tn) {
      const int col  = n0 + wn * 32 + tn * 16 + ln;
      const float bb = b_enc[col];
      #pragma unroll
      for (int r = 0; r < 8; ++r) {
        const int row = m0 + wm * 64 + tm * 16 + half * 8 + r;
        float v = acc[tm][tn][r] + bb;
        z[(size_t)row * DSAE + col] = v > 0.0f ? v : 0.0f;
      }
    }
  }
}

// ---------------------------------------------------------------------------
// Kernel 1 (fallback): fp32 inputs, convert-at-staging.
// ---------------------------------------------------------------------------
__global__ __launch_bounds__(256)
void sae_encode_wmma(const float* __restrict__ x,
                     const float* __restrict__ W_enc,
                     const float* __restrict__ b_enc,
                     float* __restrict__ z) {
  __shared__ __align__(16) __bf16 Al[BM * LDA];
  __shared__ __align__(16) __bf16 Bl[BN * LDB];

  const int tid  = threadIdx.x;
  const int lane = tid & 31;
  const int wid  = tid >> 5;
  const int wm   = wid >> 2;
  const int wn   = wid & 3;
  const int half = lane >> 4;
  const int ln   = lane & 15;
  const int m0   = blockIdx.y * BM;
  const int n0   = blockIdx.x * BN;

  v8f acc[4][2];
  #pragma unroll
  for (int i = 0; i < 4; ++i)
    #pragma unroll
    for (int j = 0; j < 2; ++j)
      #pragma unroll
      for (int e = 0; e < 8; ++e) acc[i][j][e] = 0.0f;

  for (int k0 = 0; k0 < DMODEL; k0 += BK) {
    #pragma unroll
    for (int it = 0; it < 4; ++it) {
      int flat = tid + it * 256;
      int r    = flat >> 3;
      int kc   = (flat & 7) << 2;
      const float4 f = *(const float4*)(x + (size_t)(m0 + r) * DMODEL + k0 + kc);
      union { __bf16 h[4]; uint64_t u; } p;
      p.h[0] = cvt_bf16(f.x); p.h[1] = cvt_bf16(f.y);
      p.h[2] = cvt_bf16(f.z); p.h[3] = cvt_bf16(f.w);
      *(uint64_t*)&Al[r * LDA + kc] = p.u;
    }
    #pragma unroll
    for (int it = 0; it < 4; ++it) {
      int flat = tid + it * 256;
      int kr   = flat >> 5;
      int nc   = (flat & 31) << 2;
      const float4 f = *(const float4*)(W_enc + (size_t)(k0 + kr) * DSAE + n0 + nc);
      Bl[(nc + 0) * LDB + kr] = cvt_bf16(f.x);
      Bl[(nc + 1) * LDB + kr] = cvt_bf16(f.y);
      Bl[(nc + 2) * LDB + kr] = cvt_bf16(f.z);
      Bl[(nc + 3) * LDB + kr] = cvt_bf16(f.w);
    }
    if (k0 + BK < DMODEL) {
      int r = tid >> 3, kc = (tid & 7) << 2;
      __builtin_prefetch(x + (size_t)(m0 + r) * DMODEL + (k0 + BK) + kc, 0, 3);
      int kr = tid >> 5, nc = (tid & 31) << 2;
      __builtin_prefetch(W_enc + (size_t)(k0 + BK + kr) * DSAE + n0 + nc, 0, 3);
    }
    __syncthreads();

    FragAB fa[4], fb[2];
    #pragma unroll
    for (int tm = 0; tm < 4; ++tm) {
      const __bf16* ap = &Al[(wm * 64 + tm * 16 + ln) * LDA + (half << 3)];
      fa[tm].h[0] = *(const v8bf*)ap;
      fa[tm].h[1] = *(const v8bf*)(ap + 16);
    }
    #pragma unroll
    for (int tn = 0; tn < 2; ++tn) {
      const __bf16* bp = &Bl[(wn * 32 + tn * 16 + ln) * LDB + (half << 3)];
      fb[tn].h[0] = *(const v8bf*)bp;
      fb[tn].h[1] = *(const v8bf*)(bp + 16);
    }
    #pragma unroll
    for (int tm = 0; tm < 4; ++tm)
      #pragma unroll
      for (int tn = 0; tn < 2; ++tn)
        acc[tm][tn] = __builtin_amdgcn_wmma_f32_16x16x32_bf16(
            false, fa[tm].v, false, fb[tn].v, (short)0, acc[tm][tn], false, false);
    __syncthreads();
  }

  #pragma unroll
  for (int tm = 0; tm < 4; ++tm) {
    #pragma unroll
    for (int tn = 0; tn < 2; ++tn) {
      const int col  = n0 + wn * 32 + tn * 16 + ln;
      const float bb = b_enc[col];
      #pragma unroll
      for (int r = 0; r < 8; ++r) {
        const int row = m0 + wm * 64 + tm * 16 + half * 8 + r;
        float v = acc[tm][tn][r] + bb;
        z[(size_t)row * DSAE + col] = v > 0.0f ? v : 0.0f;
      }
    }
  }
}

// ---------------------------------------------------------------------------
// Kernel 2: exact per-row top-64 via 4-round byte radix select (regs-resident).
// ---------------------------------------------------------------------------
__global__ __launch_bounds__(256)
void sae_topk(float* __restrict__ z,
              float* __restrict__ wvals, int* __restrict__ widx, int use_ws) {
  const int row = blockIdx.x;
  const int tid = threadIdx.x;
  float* zr = z + (size_t)row * DSAE;

  __shared__ uint32_t hist[256];
  __shared__ uint32_t ssum[256];
  __shared__ uint32_t s_bin, s_kneed, s_eqc, s_outc;

  uint32_t rv[64];
  #pragma unroll
  for (int j = 0; j < 64; ++j) {
    union { float f; uint32_t u; } a; a.f = zr[tid + 256 * j];
    rv[j] = a.u;
  }

  uint32_t prefix = 0, mask = 0, kneed = KTOP;
  #pragma unroll
  for (int shift = 24; shift >= 0; shift -= 8) {
    hist[tid] = 0;
    __syncthreads();
    #pragma unroll
    for (int j = 0; j < 64; ++j) {
      uint32_t v = rv[j];
      if ((v & mask) == prefix) atomicAdd(&hist[(v >> shift) & 255u], 1u);
    }
    __syncthreads();
    ssum[tid] = hist[tid];
    __syncthreads();
    for (int off = 1; off < 256; off <<= 1) {
      uint32_t add = (tid + off < 256) ? ssum[tid + off] : 0u;
      __syncthreads();
      ssum[tid] += add;
      __syncthreads();
    }
    uint32_t above = (tid == 255) ? 0u : ssum[tid + 1];
    if (ssum[tid] >= kneed && above < kneed) { s_bin = (uint32_t)tid; s_kneed = kneed - above; }
    __syncthreads();
    prefix |= (s_bin << shift);
    mask   |= (255u << shift);
    kneed   = s_kneed;
    __syncthreads();
  }
  const uint32_t T = prefix;
  if (tid == 0) { s_eqc = 0; s_outc = 0; }
  __syncthreads();

  #pragma unroll
  for (int j = 0; j < 64; ++j) {
    uint32_t v = rv[j];
    bool keep = v > T;
    if (!keep && v == T) keep = (atomicAdd(&s_eqc, 1u) < kneed);
    union { uint32_t u; float f; } a; a.u = keep ? v : 0u;
    zr[tid + 256 * j] = a.f;
    if (keep && use_ws) {
      uint32_t p = atomicAdd(&s_outc, 1u);
      wvals[(size_t)row * KTOP + p] = a.f;
      widx [(size_t)row * KTOP + p] = tid + 256 * j;
    }
  }
}

// ---------------------------------------------------------------------------
// Kernel 3: recon = z @ W_dec + b_dec (64 nnz/row; W_dec L2-resident).
// ---------------------------------------------------------------------------
__global__ __launch_bounds__(256)
void sae_decode(const float* __restrict__ z,
                const float* __restrict__ wvals, const int* __restrict__ widx,
                const float* __restrict__ W_dec, const float* __restrict__ b_dec,
                float* __restrict__ recon, int use_ws) {
  const int row = blockIdx.x;
  const int tid = threadIdx.x;
  __shared__ float sval[KTOP];
  __shared__ int   sidx[KTOP];
  __shared__ uint32_t s_cnt;

  if (use_ws) {
    if (tid < KTOP) {
      sval[tid] = wvals[(size_t)row * KTOP + tid];
      sidx[tid] = widx [(size_t)row * KTOP + tid];
    }
    if (tid == 0) s_cnt = KTOP;
    __syncthreads();
  } else {
    if (tid == 0) s_cnt = 0;
    __syncthreads();
    const float* zr = z + (size_t)row * DSAE;
    for (int i = tid; i < DSAE; i += 256) {
      float v = zr[i];
      if (v != 0.0f) {
        uint32_t p = atomicAdd(&s_cnt, 1u);
        if (p < KTOP) { sval[p] = v; sidx[p] = i; }
      }
    }
    __syncthreads();
  }

  const int c0 = tid, c1 = tid + 256, c2 = tid + 512;
  float a0 = 0.f, a1 = 0.f, a2 = 0.f;
  const int n = (s_cnt < (uint32_t)KTOP) ? (int)s_cnt : KTOP;
  for (int j = 0; j < n; ++j) {
    const float v = sval[j];
    const float* wr = W_dec + (size_t)sidx[j] * DMODEL;
    a0 += v * wr[c0];
    a1 += v * wr[c1];
    a2 += v * wr[c2];
  }
  float* rr = recon + (size_t)row * DMODEL;
  rr[c0] = a0 + b_dec[c0];
  rr[c1] = a1 + b_dec[c1];
  rr[c2] = a2 + b_dec[c2];
}

// ---------------------------------------------------------------------------
extern "C" void kernel_launch(void* const* d_in, const int* in_sizes, int n_in,
                              void* d_out, int out_size, void* d_ws, size_t ws_size,
                              hipStream_t stream) {
  (void)in_sizes; (void)n_in; (void)out_size;
  const float* x     = (const float*)d_in[0];
  const float* W_enc = (const float*)d_in[1];
  const float* b_enc = (const float*)d_in[2];
  const float* W_dec = (const float*)d_in[3];
  const float* b_dec = (const float*)d_in[4];
  // d_in[5] is k (== 64), baked in as KTOP.

  float* z     = (float*)d_out;
  float* recon = z + (size_t)BATCH * DSAE;

  // ws layout: [vals 2MB][idx 2MB][xb bf16 12.6MB][wt bf16 25.2MB]
  const size_t need_lists = (size_t)BATCH * KTOP * 8;
  const size_t xb_bytes   = (size_t)BATCH * DMODEL * 2;
  const size_t wt_bytes   = (size_t)DSAE * DMODEL * 2;
  const size_t need_full  = need_lists + xb_bytes + wt_bytes;

  const int use_ws = (ws_size >= need_lists) ? 1 : 0;
  float*    wvals = (float*)d_ws;
  int*      widx  = (int*)((char*)d_ws + (size_t)BATCH * KTOP * 4);
  uint16_t* xb    = (uint16_t*)((char*)d_ws + need_lists);
  uint16_t* wt    = (uint16_t*)((char*)d_ws + need_lists + xb_bytes);

  dim3 gEnc(DSAE / BN, BATCH / BM);   // 128 x 64 blocks

  if (ws_size >= need_full) {
    k_cvt_x<<<(BATCH * DMODEL) / (256 * 8), 256, 0, stream>>>(x, xb);
    k_cvt_wT<<<dim3(DSAE / 64, DMODEL / 64), 256, 0, stream>>>(W_enc, wt);
    sae_encode_wmma_async<<<gEnc, 256, 0, stream>>>(xb, wt, b_enc, z);
  } else {
    sae_encode_wmma<<<gEnc, 256, 0, stream>>>(x, W_enc, b_enc, z);
  }
  sae_topk<<<BATCH, 256, 0, stream>>>(z, wvals, widx, use_ws);
  sae_decode<<<BATCH, 256, 0, stream>>>(z, wvals, widx, W_dec, b_dec, recon, use_ws);
}
